// GraphRecurModule_8916352106977
// MI455X (gfx1250) — matrix operand order repaired and checked by reading, not compiled, not run on hardware
//
#include <hip/hip_runtime.h>
#include <hip/hip_bf16.h>
#include <stdint.h>

typedef __bf16 bf16_t;
typedef __attribute__((ext_vector_type(16))) __bf16 v16bf;
typedef __attribute__((ext_vector_type(8)))  __bf16 v8bf;
typedef __attribute__((ext_vector_type(8)))  float  v8f;
typedef __attribute__((ext_vector_type(4)))  int    v4i;

#define BM 256
#define BN 64
#define BK 32
#define LDSK 40   // padded LDS row stride (elements); 80B rows stay 16B-aligned, bank-skewed

typedef __attribute__((address_space(1))) v4i* gptr_v4i;
typedef __attribute__((address_space(3))) v4i* lptr_v4i;

#if defined(__gfx1250__) && __has_builtin(__builtin_amdgcn_global_load_async_to_lds_b128)
#define USE_ASYNC_LDS 1
#else
#define USE_ASYNC_LDS 0
#endif

// Stage 16B global -> LDS. Prefers the CDNA5 async LDS-DMA path (ASYNCcnt).
__device__ __forceinline__ void stage16(const bf16_t* g, bf16_t* l) {
#if USE_ASYNC_LDS
  __builtin_amdgcn_global_load_async_to_lds_b128(
      (gptr_v4i)(v4i*)g, (lptr_v4i)(v4i*)l, 0, 0);
#else
  *(v8bf*)l = *(const v8bf*)g;
#endif
}

__device__ __forceinline__ void async_wait_all() {
#if USE_ASYNC_LDS
#if __has_builtin(__builtin_amdgcn_s_wait_asynccnt)
  __builtin_amdgcn_s_wait_asynccnt(0);
#else
  asm volatile("s_wait_asynccnt 0" ::: "memory");
#endif
#endif
}

__device__ __forceinline__ v16bf mk16(v8bf lo, v8bf hi) {
  union { v16bf v; v8bf h[2]; } u;
  u.h[0] = lo; u.h[1] = hi;
  return u.v;
}

__device__ __forceinline__ v8f v8f_zero() {
  v8f z = {0.f, 0.f, 0.f, 0.f, 0.f, 0.f, 0.f, 0.f};
  return z;
}

// One BK=32 K-step: 8 waves, wave wm owns rows [wm*32, wm*32+32) x all 64 cols
// as 2x4 v_wmma_f32_16x16x32_bf16. A: LDS row-major [m][k]; B: LDS transposed [n][k].
__device__ __forceinline__ void wmma_step(const bf16_t As[BM][LDSK],
                                          const bf16_t Bs[BN][LDSK],
                                          int wm, int lane, v8f acc[2][4]) {
  const int row16 = lane & 15;
  const int hi    = lane >> 4;
  const int klA   = hi * 16;   // A frag: lanes0-15 k0..15, lanes16-31 k16..31
  const int ksB   = hi * 16;   // B frag: lanes0-15 k0..15, lanes16-31 k16..31
  v16bf a[2], b[4];
#pragma unroll
  for (int i = 0; i < 2; ++i) {
    int r = wm * 32 + i * 16 + row16;
    a[i] = mk16(*(const v8bf*)&As[r][klA], *(const v8bf*)&As[r][klA + 8]);
  }
#pragma unroll
  for (int j = 0; j < 4; ++j) {
    int c = j * 16 + row16;
    b[j] = mk16(*(const v8bf*)&Bs[c][ksB], *(const v8bf*)&Bs[c][ksB + 8]);
  }
#pragma unroll
  for (int i = 0; i < 2; ++i)
#pragma unroll
    for (int j = 0; j < 4; ++j)
      acc[i][j] = __builtin_amdgcn_wmma_f32_16x16x32_bf16(
          false, a[i], false, b[j], (short)0, acc[i][j], false, false);
}

// ---------------- prep kernels ----------------

__global__ void k_rowdeg(const float* __restrict__ adj, float* __restrict__ dis) {
  const int i = blockIdx.x;
  __shared__ float red[256];
  float s = 0.f;
  for (int j = threadIdx.x; j < 2048; j += 256)
    if (j != i) s += adj[(size_t)i * 2048 + j];
  red[threadIdx.x] = s;
  __syncthreads();
  for (int st = 128; st > 0; st >>= 1) {
    if (threadIdx.x < st) red[threadIdx.x] += red[threadIdx.x + st];
    __syncthreads();
  }
  if (threadIdx.x == 0) {
    float d = red[0];
    dis[i] = d > 0.f ? rsqrtf(d) : 0.f;
  }
}

__global__ void k_build_S(const float* __restrict__ adj, const float* __restrict__ dis,
                          bf16_t* __restrict__ S) {
  size_t idx = (size_t)blockIdx.x * 256 + threadIdx.x;
  int i = (int)(idx >> 11), j = (int)(idx & 2047);
  float v = (i == j) ? 0.f : -dis[i] * adj[idx] * dis[j];
  S[idx] = (bf16_t)v;
}

__global__ void k_cvt_bf16(const float* __restrict__ in, bf16_t* __restrict__ out, size_t n) {
  for (size_t i = (size_t)blockIdx.x * blockDim.x + threadIdx.x; i < n;
       i += (size_t)gridDim.x * blockDim.x)
    out[i] = (bf16_t)in[i];
}

// WT[k][g] (256x512 row-major bf16) = k<128 ? Wih[g][k] : Whh[g][k-128]
__global__ void k_build_wcat(const float* __restrict__ Wih, const float* __restrict__ Whh,
                             bf16_t* __restrict__ WT) {
  int idx = blockIdx.x * 256 + threadIdx.x;
  int k = idx >> 9, g = idx & 511;
  float v = (k < 128) ? Wih[g * 128 + k] : Whh[g * 128 + (k - 128)];
  WT[(size_t)k * 512 + g] = (bf16_t)v;
}

__global__ void k_zero_u32(uint32_t* __restrict__ p, size_t n) {
  for (size_t i = (size_t)blockIdx.x * blockDim.x + threadIdx.x; i < n;
       i += (size_t)gridDim.x * blockDim.x)
    p[i] = 0u;
}

// ---------------- GEMM kernels ----------------

// Y[t] = S @ X[t] ; S (2048x2048 bf16), X (NT,2048,Dc), Y (NT,2048,Dc) bf16
__global__ void __launch_bounds__(256) k_gemm_S_X(
    const bf16_t* __restrict__ S, const bf16_t* __restrict__ X,
    bf16_t* __restrict__ Y, int K, int Dc) {
  const int t = blockIdx.z;
  const bf16_t* Xt = X + (size_t)t * K * Dc;
  bf16_t* Yt = Y + (size_t)t * 2048 * Dc;
  const int m0 = blockIdx.x * BM;
  const int n0 = blockIdx.y * BN;

  __shared__ bf16_t As[BM][LDSK];
  __shared__ bf16_t Bs[BN][LDSK];

  const int tid = threadIdx.x, lane = tid & 31, wm = tid >> 5;

  // A staging: 4 fixed 16B groups per thread, pointer-advance per K-step
  const bf16_t* pA[4];
  bf16_t* lA[4];
#pragma unroll
  for (int q = 0; q < 4; ++q) {
    int g = tid + q * 256;
    int r = g >> 2, kc = (g & 3) * 8;
    pA[q] = S + (size_t)(m0 + r) * K + kc;
    lA[q] = &As[r][kc];
  }
  // B staging: one 16B group per thread, stored transposed
  const int br = tid >> 3, bnc = (tid & 7) * 8;
  const bf16_t* pB = Xt + (size_t)br * Dc + n0 + bnc;

  v8f acc[2][4];
#pragma unroll
  for (int i = 0; i < 2; ++i)
#pragma unroll
    for (int j = 0; j < 4; ++j) acc[i][j] = v8f_zero();

  for (int kb = 0; kb < K; kb += BK) {
#pragma unroll
    for (int q = 0; q < 4; ++q) {
      stage16(pA[q], lA[q]);
      __builtin_prefetch((const void*)(pA[q] + 2 * BK), 0, 1);  // next-next S slab
      pA[q] += BK;
    }
    v8bf bv = *(const v8bf*)pB;
    pB += (size_t)BK * Dc;
#pragma unroll
    for (int q = 0; q < 8; ++q) Bs[bnc + q][br] = bv[q];
    async_wait_all();
    __syncthreads();
    wmma_step(As, Bs, wm, lane, acc);
    __syncthreads();
  }

  const int row16 = lane & 15, mh = (lane >> 4) * 8;
#pragma unroll
  for (int i = 0; i < 2; ++i)
#pragma unroll
    for (int j = 0; j < 4; ++j) {
      int nn = n0 + j * 16 + row16;
#pragma unroll
      for (int v = 0; v < 8; ++v) {
        int m = m0 + wm * 32 + i * 16 + mh + v;
        Yt[(size_t)m * Dc + nn] = (bf16_t)acc[i][j][v];
      }
    }
}

// out = gelu(Xa@W0 + Ya@W1 + b) ; Xa,Ya (Mtot x Dc) bf16; W (Dc x 128) bf16; out bf16
__global__ void __launch_bounds__(256) k_cheb_combine(
    const bf16_t* __restrict__ Xa, const bf16_t* __restrict__ Ya,
    const bf16_t* __restrict__ W0, const bf16_t* __restrict__ W1,
    const float* __restrict__ bias, bf16_t* __restrict__ outp, int Dc) {
  const int m0 = blockIdx.x * BM;
  const int n0 = blockIdx.y * BN;

  __shared__ bf16_t As[BM][LDSK];
  __shared__ bf16_t Bs[BN][LDSK];

  const int tid = threadIdx.x, lane = tid & 31, wm = tid >> 5;
  const int br = tid >> 3, bnc = (tid & 7) * 8;

  int rq[4], kcq[4];
  bf16_t* lA[4];
#pragma unroll
  for (int q = 0; q < 4; ++q) {
    int g = tid + q * 256;
    rq[q] = g >> 2;
    kcq[q] = (g & 3) * 8;
    lA[q] = &As[rq[q]][kcq[q]];
  }

  v8f acc[2][4];
#pragma unroll
  for (int i = 0; i < 2; ++i)
#pragma unroll
    for (int j = 0; j < 4; ++j) acc[i][j] = v8f_zero();

#pragma unroll
  for (int p = 0; p < 2; ++p) {
    const bf16_t* A = p ? Ya : Xa;
    const bf16_t* W = p ? W1 : W0;
    const bf16_t* pA[4];
#pragma unroll
    for (int q = 0; q < 4; ++q) pA[q] = A + (size_t)(m0 + rq[q]) * Dc + kcq[q];
    const bf16_t* pB = W + (size_t)br * 128 + n0 + bnc;

    for (int kb = 0; kb < Dc; kb += BK) {
#pragma unroll
      for (int q = 0; q < 4; ++q) {
        stage16(pA[q], lA[q]);
        pA[q] += BK;
      }
      v8bf bv = *(const v8bf*)pB;
      pB += (size_t)BK * 128;
#pragma unroll
      for (int q2 = 0; q2 < 8; ++q2) Bs[bnc + q2][br] = bv[q2];
      async_wait_all();
      __syncthreads();
      wmma_step(As, Bs, wm, lane, acc);
      __syncthreads();
    }
  }

  const int row16 = lane & 15, mh = (lane >> 4) * 8;
#pragma unroll
  for (int i = 0; i < 2; ++i)
#pragma unroll
    for (int j = 0; j < 4; ++j) {
      int nn = n0 + j * 16 + row16;
      float bv = bias[nn];
#pragma unroll
      for (int v = 0; v < 8; ++v) {
        int m = m0 + wm * 32 + i * 16 + mh + v;
        float x = acc[i][j][v] + bv;
        float ge = 0.5f * x * (1.0f + erff(x * 0.70710678118654752f));
        outp[(size_t)m * 128 + nn] = (bf16_t)ge;
      }
    }
}

// gates = [x_t | h_prev] @ WT ; M=8192, N=512, K=256; gates f32
// mode 0: x_t row m is act1[((m>>11)*T + t)*B + (m&2047)]
// mode 1: x_t row m is hseq[(t*8192 + m)]
__global__ void __launch_bounds__(256) k_lstm_gemm(
    const bf16_t* __restrict__ xsrc, int mode, int t,
    const bf16_t* __restrict__ hprev, const bf16_t* __restrict__ WT,
    float* __restrict__ gates) {
  const int m0 = blockIdx.x * BM;
  const int n0 = blockIdx.y * BN;

  __shared__ bf16_t As[BM][LDSK];
  __shared__ bf16_t Bs[BN][LDSK];

  const int tid = threadIdx.x, lane = tid & 31, wm = tid >> 5;
  const int br = tid >> 3, bnc = (tid & 7) * 8;

  const bf16_t* px[4];
  const bf16_t* phh[4];
  bf16_t* lA[4];
#pragma unroll
  for (int q = 0; q < 4; ++q) {
    int g = tid + q * 256;
    int r = g >> 2, kc = (g & 3) * 8;
    int m = m0 + r;
    size_t rowoff = (mode == 0)
        ? ((size_t)((m >> 11) * 64 + t) * 2048 + (size_t)(m & 2047)) * 128
        : ((size_t)t * 8192 + (size_t)m) * 128;
    px[q]  = xsrc + rowoff + kc;
    phh[q] = hprev + (size_t)m * 128 + kc;
    lA[q]  = &As[r][kc];
  }
  const bf16_t* pB = WT + (size_t)br * 512 + n0 + bnc;

  v8f acc[2][4];
#pragma unroll
  for (int i = 0; i < 2; ++i)
#pragma unroll
    for (int j = 0; j < 4; ++j) acc[i][j] = v8f_zero();

#pragma unroll
  for (int kb = 0; kb < 256; kb += BK) {   // fully unrolled: x/h branch folds
#pragma unroll
    for (int q = 0; q < 4; ++q) {
      const bf16_t* src = (kb < 128) ? (px[q] + kb) : (phh[q] + (kb - 128));
      stage16(src, lA[q]);
    }
    v8bf bv = *(const v8bf*)pB;
    pB += (size_t)BK * 512;
#pragma unroll
    for (int q = 0; q < 8; ++q) Bs[bnc + q][br] = bv[q];
    async_wait_all();
    __syncthreads();
    wmma_step(As, Bs, wm, lane, acc);
    __syncthreads();
  }

  const int row16 = lane & 15, mh = (lane >> 4) * 8;
#pragma unroll
  for (int i = 0; i < 2; ++i)
#pragma unroll
    for (int j = 0; j < 4; ++j) {
      int nn = n0 + j * 16 + row16;
#pragma unroll
      for (int v = 0; v < 8; ++v) {
        int m = m0 + wm * 32 + i * 16 + mh + v;
        gates[(size_t)m * 512 + nn] = acc[i][j][v];
      }
    }
}

// LSTM cell pointwise update. 8192*128 threads.
__global__ void k_lstm_update(const float* __restrict__ gates,
                              const float* __restrict__ bih, const float* __restrict__ bhh,
                              float* __restrict__ c, bf16_t* __restrict__ h,
                              bf16_t* __restrict__ hseq_dst, float* __restrict__ out_dst,
                              int t) {
  const int idx = blockIdx.x * 256 + threadIdx.x;   // m*128 + j
  const int m = idx >> 7, j = idx & 127;
  const size_t gb = (size_t)m * 512 + j;
  float gi = gates[gb]       + bih[j]       + bhh[j];
  float gf = gates[gb + 128] + bih[128 + j] + bhh[128 + j];
  float gg = gates[gb + 256] + bih[256 + j] + bhh[256 + j];
  float go = gates[gb + 384] + bih[384 + j] + bhh[384 + j];
  float i_ = 1.f / (1.f + __expf(-gi));
  float f_ = 1.f / (1.f + __expf(-gf));
  float g_ = tanhf(gg);
  float o_ = 1.f / (1.f + __expf(-go));
  float cn = f_ * c[idx] + i_ * g_;
  float hn = o_ * tanhf(cn);
  c[idx] = cn;
  h[idx] = (bf16_t)hn;
  if (hseq_dst) hseq_dst[idx] = (bf16_t)hn;
  if (out_dst) {
    int n = m >> 11, b = m & 2047;
    out_dst[(((size_t)n * 64 + t) * 2048 + b) * 128 + j] = hn;
  }
}

// ---------------- launcher ----------------

extern "C" void kernel_launch(void* const* d_in, const int* in_sizes, int n_in,
                              void* d_out, int out_size, void* d_ws, size_t ws_size,
                              hipStream_t stream) {
  (void)in_sizes; (void)n_in; (void)out_size; (void)ws_size;
  const float* input = (const float*)d_in[0];
  const float* adj   = (const float*)d_in[1];
  const float* W0_0  = (const float*)d_in[2];
  const float* W1_0  = (const float*)d_in[3];
  const float* b_0   = (const float*)d_in[4];
  const float* W0_1  = (const float*)d_in[5];
  const float* W1_1  = (const float*)d_in[6];
  const float* b_1   = (const float*)d_in[7];
  const float* Wih0  = (const float*)d_in[8];
  const float* Whh0  = (const float*)d_in[9];
  const float* bih0  = (const float*)d_in[10];
  const float* bhh0  = (const float*)d_in[11];
  const float* Wih1  = (const float*)d_in[12];
  const float* Whh1  = (const float*)d_in[13];
  const float* bih1  = (const float*)d_in[14];
  const float* bhh1  = (const float*)d_in[15];
  float* out = (float*)d_out;

  char* ws = (char*)d_ws;
  size_t off = 0;
  auto take = [&](size_t bytes) {
    char* p = ws + off;
    off = (off + bytes + 255) & ~(size_t)255;
    return p;
  };
  bf16_t* Sbf   = (bf16_t*)take((size_t)2048 * 2048 * 2);      // 8 MB
  float*  dis   = (float*) take(2048 * 4);
  bf16_t* x0bf  = (bf16_t*)take((size_t)256 * 2048 * 64 * 2);  // 64 MB
  bf16_t* Ybf   = (bf16_t*)take((size_t)256 * 2048 * 128 * 2); // 128 MB
  bf16_t* act0  = (bf16_t*)take((size_t)256 * 2048 * 128 * 2); // 128 MB, reused as hseq0
  bf16_t* act1  = (bf16_t*)take((size_t)256 * 2048 * 128 * 2); // 128 MB
  bf16_t* W00b  = (bf16_t*)take(64 * 128 * 2);
  bf16_t* W10b  = (bf16_t*)take(64 * 128 * 2);
  bf16_t* W01b  = (bf16_t*)take(128 * 128 * 2);
  bf16_t* W11b  = (bf16_t*)take(128 * 128 * 2);
  bf16_t* Wcat0 = (bf16_t*)take(256 * 512 * 2);
  bf16_t* Wcat1 = (bf16_t*)take(256 * 512 * 2);
  float*  gates = (float*) take((size_t)8192 * 512 * 4);       // 16 MB
  const size_t hbytes = (size_t)8192 * 128 * 2;
  const size_t cbytes = (size_t)8192 * 128 * 4;
  char* hc = take(2 * (hbytes + cbytes));
  bf16_t* h0 = (bf16_t*)hc;
  float*  c0 = (float*)(hc + hbytes);
  bf16_t* h1 = (bf16_t*)(hc + hbytes + cbytes);
  float*  c1 = (float*)(hc + 2 * hbytes + cbytes);
  bf16_t* hseq0 = act0;  // act0 is dead after cheb layer 1

  // prep
  k_rowdeg<<<2048, 256, 0, stream>>>(adj, dis);
  k_build_S<<<(2048 * 2048) / 256, 256, 0, stream>>>(adj, dis, Sbf);
  k_cvt_bf16<<<4096, 256, 0, stream>>>(input, x0bf, (size_t)256 * 2048 * 64);
  k_cvt_bf16<<<32, 256, 0, stream>>>(W0_0, W00b, 64 * 128);
  k_cvt_bf16<<<32, 256, 0, stream>>>(W1_0, W10b, 64 * 128);
  k_cvt_bf16<<<64, 256, 0, stream>>>(W0_1, W01b, 128 * 128);
  k_cvt_bf16<<<64, 256, 0, stream>>>(W1_1, W11b, 128 * 128);
  k_build_wcat<<<512, 256, 0, stream>>>(Wih0, Whh0, Wcat0);
  k_build_wcat<<<512, 256, 0, stream>>>(Wih1, Whh1, Wcat1);

  // Cheb layer 0: Y = S@x ; act0 = gelu(x@W0 + Y@W1 + b)
  k_gemm_S_X<<<dim3(8, 1, 256), 256, 0, stream>>>(Sbf, x0bf, Ybf, 2048, 64);
  k_cheb_combine<<<dim3(2048, 2), 256, 0, stream>>>(x0bf, Ybf, W00b, W10b, b_0, act0, 64);
  // Cheb layer 1
  k_gemm_S_X<<<dim3(8, 2, 256), 256, 0, stream>>>(Sbf, act0, Ybf, 2048, 128);
  k_cheb_combine<<<dim3(2048, 2), 256, 0, stream>>>(act0, Ybf, W01b, W11b, b_1, act1, 128);

  // zero h0/c0/h1/c1
  k_zero_u32<<<4096, 256, 0, stream>>>((uint32_t*)hc, 2 * (hbytes + cbytes) / 4);

  // LSTM layer 0 over act1, hidden sequence -> hseq0 (bf16)
  for (int t = 0; t < 64; ++t) {
    k_lstm_gemm<<<dim3(32, 8), 256, 0, stream>>>(act1, 0, t, h0, Wcat0, gates);
    k_lstm_update<<<4096, 256, 0, stream>>>(gates, bih0, bhh0, c0, h0,
                                            hseq0 + (size_t)t * 8192 * 128,
                                            (float*)nullptr, t);
  }
  // LSTM layer 1 over hseq0, hidden -> d_out (f32, (N,T,B,H) layout)
  for (int t = 0; t < 64; ++t) {
    k_lstm_gemm<<<dim3(32, 8), 256, 0, stream>>>(hseq0, 1, t, h1, Wcat1, gates);
    k_lstm_update<<<4096, 256, 0, stream>>>(gates, bih1, bhh1, c1, h1,
                                            (bf16_t*)nullptr, out, t);
  }
}
